// SAGE_edge_26843545600712
// MI455X (gfx1250) — matrix operand order repaired and checked by reading, not compiled, not run on hardware
//
#include <hip/hip_runtime.h>

// ---------------------------------------------------------------------------
// Problem constants (match reference)
// ---------------------------------------------------------------------------
#define NN   10000
#define EE   320000
#define INC  128
#define HIDC 256
#define OUTC 128
#define GG   64

// ---------------------------------------------------------------------------
// Types for WMMA (wave32, gfx1250)
// ---------------------------------------------------------------------------
typedef __attribute__((ext_vector_type(16))) __bf16        v16bf;
typedef __attribute__((ext_vector_type(8)))  float         v8f;
typedef __attribute__((ext_vector_type(4)))  unsigned int  u32x4;
typedef __attribute__((ext_vector_type(4)))  float         f32x4;

union FragU {
    v16bf          v;
    u32x4          q[2];
    unsigned short us[16];
};

__device__ __forceinline__ unsigned short f2bf(float x) {
    unsigned int u = __float_as_uint(x);
    u += 0x7FFFu + ((u >> 16) & 1u);            // round-to-nearest-even
    return (unsigned short)(u >> 16);
}

__device__ __forceinline__ void atomAddF(float* p, float v) {
    unsafeAtomicAdd(p, v);                      // native global_atomic_add_f32
}

// ---------------------------------------------------------------------------
// fp32 -> bf16 conversion (activations)
// ---------------------------------------------------------------------------
__global__ void cvt_bf_kernel(const float* __restrict__ in,
                              unsigned short* __restrict__ out, int n) {
    int i = blockIdx.x * blockDim.x + threadIdx.x;
    if (i < n) out[i] = f2bf(in[i]);
}

// ---------------------------------------------------------------------------
// Pack fp32 weights B[K,Nc] into WMMA-B fragment order (bf16):
//   Bp[((tn*(K/32) + kb)*32 + lane)*16 + idx]  with lane = half*16 + l,
//   holds B[kb*32 + half*16 + idx][tn*16 + l].
// One thread per packed element. Fuses f32->bf16 conversion.
// ---------------------------------------------------------------------------
__global__ void pack_b_kernel(const float* __restrict__ w,
                              unsigned short* __restrict__ bp, int K, int Nc) {
    int i = blockIdx.x * blockDim.x + threadIdx.x;
    if (i >= K * Nc) return;
    int idx  = i & 15;
    int lane = (i >> 4) & 31;
    int blk  = i >> 9;                 // tn*(K/32) + kb
    int kblk = K >> 5;
    int kb   = blk % kblk;
    int tn   = blk / kblk;
    int half = lane >> 4;
    int l    = lane & 15;
    int k    = kb * 32 + half * 16 + idx;
    int col  = tn * 16 + l;
    bp[i] = f2bf(w[(size_t)k * Nc + col]);
}

// ---------------------------------------------------------------------------
// WMMA GEMM:  O[M,Nc] = act( A_bf16[M,K] @ B[K,Nc] + bias + addsrc )
// B is prepacked (pack_b_kernel). One wave computes a 16x64 strip (4 n-tiles),
// reusing each A fragment across 4 WMMAs. K templated -> fully unrolled.
// ACT: 0 = none, 1 = relu, 2 = leaky(0.01)
// ---------------------------------------------------------------------------
template <int K, int ACT, bool HAS_BIAS, bool HAS_ADD, bool HAS_OBF>
__global__ void __launch_bounds__(256)
wmma_gemm_bf16(const unsigned short* __restrict__ A,
               const unsigned short* __restrict__ Bp,
               const float* __restrict__ bias,
               const float* __restrict__ addsrc,
               float* __restrict__ out,
               unsigned short* __restrict__ outbf,
               int M, int Nc)
{
    constexpr int KB = K / 32;
    int wave  = (int)((blockIdx.x * blockDim.x + threadIdx.x) >> 5);
    int ngrp  = Nc >> 6;                        // groups of 4 n-tiles
    int total = (M >> 4) * ngrp;
    if (wave >= total) return;                  // wave-uniform guard: EXEC all-1s

    int tm = wave / ngrp;
    int tg = wave - tm * ngrp;

    int lane = threadIdx.x & 31;
    int half = lane >> 4;
    int l    = lane & 15;

    v8f acc[4] = {};

    const unsigned short* arow  = A + (size_t)(tm * 16 + l) * (size_t)K;
    const unsigned short* bbase = Bp + (size_t)(tg * 4) * KB * 512 + lane * 16;

    #pragma unroll
    for (int kb = 0; kb < KB; ++kb) {
        FragU a;
        a.q[0] = *(const u32x4*)(arow + kb * 32 + half * 16);      // K kb*32+{0..7|16..23}
        a.q[1] = *(const u32x4*)(arow + kb * 32 + half * 16 + 8);  // K kb*32+{8..15|24..31}
        #pragma unroll
        for (int t = 0; t < 4; ++t) {
            FragU b;
            const unsigned short* bp = bbase + ((size_t)t * KB + kb) * 512;
            b.q[0] = *(const u32x4*)(bp);
            b.q[1] = *(const u32x4*)(bp + 8);
            acc[t] = __builtin_amdgcn_wmma_f32_16x16x32_bf16(
                         false, a.v, false, b.v, (short)0, acc[t], false, false);
        }
    }

    #pragma unroll
    for (int t = 0; t < 4; ++t) {
        int   col  = (tg * 4 + t) * 16 + l;
        float bval = HAS_BIAS ? bias[col] : 0.0f;
        #pragma unroll
        for (int r = 0; r < 8; ++r) {
            int    row = tm * 16 + half * 8 + r;    // C/D layout: VGPR r -> M = r (+8)
            size_t o   = (size_t)row * (size_t)Nc + (size_t)col;
            float  v   = acc[t][r] + bval;
            if constexpr (HAS_ADD) v += addsrc[o];
            if constexpr (ACT == 1) v = fmaxf(v, 0.0f);
            if constexpr (ACT == 2) v = (v >= 0.0f) ? v : 0.01f * v;
            out[o] = v;
            if constexpr (HAS_OBF) outbf[o] = f2bf(v);
        }
    }
}

// ---------------------------------------------------------------------------
// Edge degree (scatter count) — dst fixed for all 3 convs, computed once/call.
// ---------------------------------------------------------------------------
__global__ void degree_kernel(const int* __restrict__ dst, float* __restrict__ deg, int E) {
    int e = blockIdx.x * blockDim.x + threadIdx.x;
    if (e < E) atomAddF(&deg[dst[e]], 1.0f);
}

// ---------------------------------------------------------------------------
// Edge message + scatter-sum:
//   m = relu(h[src] + ea[e]*We + be); agg[dst] += m   (C = 256, 4 ch/thread)
// ---------------------------------------------------------------------------
__global__ void edge_msg_kernel(const float* __restrict__ h,
                                const int* __restrict__ src,
                                const int* __restrict__ dst,
                                const float* __restrict__ ea,
                                const float* __restrict__ we,
                                const float* __restrict__ be,
                                float* __restrict__ agg, int E, int C)
{
    int idx = blockIdx.x * blockDim.x + threadIdx.x;
    int per = C >> 2;
    int e   = idx / per;
    if (e >= E) return;
    int c = (idx - e * per) << 2;
    int s = src[e], d = dst[e];
    float a = ea[e];

    f32x4 hv = *(const f32x4*)(h + (size_t)s * C + c);
    f32x4 wv = *(const f32x4*)(we + c);
    f32x4 bv = *(const f32x4*)(be + c);
    f32x4 m;
    m.x = fmaxf(hv.x + a * wv.x + bv.x, 0.0f);
    m.y = fmaxf(hv.y + a * wv.y + bv.y, 0.0f);
    m.z = fmaxf(hv.z + a * wv.z + bv.z, 0.0f);
    m.w = fmaxf(hv.w + a * wv.w + bv.w, 0.0f);

    float* o = agg + (size_t)d * C + c;
    atomAddF(o + 0, m.x);
    atomAddF(o + 1, m.y);
    atomAddF(o + 2, m.z);
    atomAddF(o + 3, m.w);
}

// ---------------------------------------------------------------------------
// agg /= max(deg,1); also emit bf16 copy for the following WMMA GEMM.
// ---------------------------------------------------------------------------
__global__ void rescale_bf_kernel(float* __restrict__ agg,
                                  const float* __restrict__ deg,
                                  unsigned short* __restrict__ obf, int N, int C)
{
    int i = blockIdx.x * blockDim.x + threadIdx.x;
    if (i >= N * C) return;
    int   row = i / C;
    float v   = agg[i] / fmaxf(deg[row], 1.0f);
    agg[i]    = v;
    obf[i]    = f2bf(v);
}

// ---------------------------------------------------------------------------
// global_mean_pool (sum phase): g[batch[i]] += h[i], bcnt[batch[i]] += 1
// ---------------------------------------------------------------------------
__global__ void pool_sum_kernel(const float* __restrict__ h,
                                const int* __restrict__ batch,
                                float* __restrict__ g, float* __restrict__ bcnt,
                                int N, int C)
{
    int idx = blockIdx.x * blockDim.x + threadIdx.x;
    int per = C >> 2;
    int i   = idx / per;
    if (i >= N) return;
    int c = (idx - i * per) << 2;
    int b = batch[i];
    f32x4 hv = *(const f32x4*)(h + (size_t)i * C + c);
    float* o = g + (size_t)b * C + c;
    atomAddF(o + 0, hv.x);
    atomAddF(o + 1, hv.y);
    atomAddF(o + 2, hv.z);
    atomAddF(o + 3, hv.w);
    if (c == 0) atomAddF(&bcnt[b], 1.0f);
}

__global__ void pool_div_kernel(float* __restrict__ g, const float* __restrict__ bcnt,
                                unsigned short* __restrict__ gbf, int G, int C)
{
    int i = blockIdx.x * blockDim.x + threadIdx.x;
    if (i >= G * C) return;
    int   row = i / C;
    float v   = g[i] / fmaxf(bcnt[row], 1.0f);
    g[i]      = v;
    gbf[i]    = f2bf(v);
}

// ---------------------------------------------------------------------------
// BatchNorm over G rows + leaky, in place on g2[G, C]. One thread per column.
// ---------------------------------------------------------------------------
__global__ void bn_leaky_kernel(float* __restrict__ g2,
                                const float* __restrict__ gamma,
                                const float* __restrict__ beta, int G, int C)
{
    int c = blockIdx.x * blockDim.x + threadIdx.x;
    if (c >= C) return;
    float mu = 0.0f;
    for (int i = 0; i < G; ++i) mu += g2[(size_t)i * C + c];
    mu /= (float)G;
    float var = 0.0f;
    for (int i = 0; i < G; ++i) {
        float d = g2[(size_t)i * C + c] - mu;
        var += d * d;
    }
    var /= (float)G;
    float inv = rsqrtf(var + 1e-5f);
    float gm = gamma[c], bt = beta[c];
    for (int i = 0; i < G; ++i) {
        float v = gm * (g2[(size_t)i * C + c] - mu) * inv + bt;
        g2[(size_t)i * C + c] = (v >= 0.0f) ? v : 0.01f * v;
    }
}

// ---------------------------------------------------------------------------
// Final head: out[g] = relu(sum_c g2[g,c] * w[c] + b)
// ---------------------------------------------------------------------------
__global__ void head_out_kernel(const float* __restrict__ g2,
                                const float* __restrict__ w,
                                const float* __restrict__ b,
                                float* __restrict__ out, int G, int C)
{
    int gi = blockIdx.x * blockDim.x + threadIdx.x;
    if (gi >= G) return;
    float s = b[0];
    for (int c = 0; c < C; ++c) s += g2[(size_t)gi * C + c] * w[c];
    out[gi] = fmaxf(s, 0.0f);
}

// ---------------------------------------------------------------------------
// Host-side GEMM launchers (explicit template instantiations)
// ---------------------------------------------------------------------------
template <int K, int ACT, bool HB, bool HA, bool HO>
static void launch_gemm(const unsigned short* A, const unsigned short* Bp,
                        const float* bias, const float* addsrc,
                        float* O, unsigned short* Obf, int M, int Nc,
                        hipStream_t s)
{
    int waves  = (M / 16) * (Nc / 64);
    int blocks = (waves + 7) / 8;               // 8 waves (256 threads) per block
    wmma_gemm_bf16<K, ACT, HB, HA, HO><<<blocks, 256, 0, s>>>(A, Bp, bias, addsrc, O, Obf, M, Nc);
}

// ---------------------------------------------------------------------------
// Host-side orchestration
// ---------------------------------------------------------------------------
extern "C" void kernel_launch(void* const* d_in, const int* in_sizes, int n_in,
                              void* d_out, int out_size, void* d_ws, size_t ws_size,
                              hipStream_t stream)
{
    (void)in_sizes; (void)n_in; (void)out_size; (void)ws_size;

    const float* x        = (const float*)d_in[0];
    const int*   ei       = (const int*)  d_in[1];
    const float* ea       = (const float*)d_in[2];
    const int*   batch    = (const int*)  d_in[3];
    const float* prelin_w = (const float*)d_in[4];
    const float* prelin_b = (const float*)d_in[5];
    const float* c1_wl    = (const float*)d_in[6];
    const float* c1_bl    = (const float*)d_in[7];
    const float* c1_wr    = (const float*)d_in[8];
    const float* c1_we    = (const float*)d_in[9];
    const float* c1_be    = (const float*)d_in[10];
    const float* c2_wl    = (const float*)d_in[11];
    const float* c2_bl    = (const float*)d_in[12];
    const float* c2_wr    = (const float*)d_in[13];
    const float* c2_we    = (const float*)d_in[14];
    const float* c2_be    = (const float*)d_in[15];
    const float* c3_wl    = (const float*)d_in[16];
    const float* c3_bl    = (const float*)d_in[17];
    const float* c3_wr    = (const float*)d_in[18];
    const float* c3_we    = (const float*)d_in[19];
    const float* c3_be    = (const float*)d_in[20];
    const float* hh1_w    = (const float*)d_in[21];
    const float* hh1_b    = (const float*)d_in[22];
    const float* hh2_w    = (const float*)d_in[23];
    const float* hh2_b    = (const float*)d_in[24];
    const float* oo_w     = (const float*)d_in[25];
    const float* oo_b     = (const float*)d_in[26];
    const float* oh_w     = (const float*)d_in[27];
    const float* oh_b     = (const float*)d_in[28];
    const float* h1_w     = (const float*)d_in[29];
    const float* h1_b     = (const float*)d_in[30];
    const float* bn_g     = (const float*)d_in[31];
    const float* bn_b     = (const float*)d_in[32];

    const int* src = ei;
    const int* dst = ei + EE;
    float* out = (float*)d_out;

    // ---- workspace carve-out (256B-aligned) ----
    char* p = (char*)d_ws;
    auto alloc = [&](size_t bytes) -> void* {
        void* r = p;
        p += (bytes + 255) & ~(size_t)255;
        return r;
    };
    float* hA  = (float*)alloc((size_t)NN * HIDC * 4);
    float* hB  = (float*)alloc((size_t)NN * HIDC * 4);
    float* tmp = (float*)alloc((size_t)NN * HIDC * 4);
    float* agg = (float*)alloc((size_t)NN * HIDC * 4);
    float* deg  = (float*)alloc((size_t)NN * 4);
    float* g    = (float*)alloc((size_t)GG * OUTC * 4);
    float* g2   = (float*)alloc((size_t)GG * HIDC * 4);
    float* bcnt = (float*)alloc((size_t)GG * 4);
    unsigned short* x_bf   = (unsigned short*)alloc((size_t)NN * INC * 2);
    unsigned short* hA_bf  = (unsigned short*)alloc((size_t)NN * HIDC * 2);
    unsigned short* hB_bf  = (unsigned short*)alloc((size_t)NN * HIDC * 2);
    unsigned short* agg_bf = (unsigned short*)alloc((size_t)NN * HIDC * 2);
    unsigned short* g_bf   = (unsigned short*)alloc((size_t)GG * OUTC * 2);
    unsigned short* w_prelin = (unsigned short*)alloc((size_t)INC * HIDC * 2);
    unsigned short* w_c1l = (unsigned short*)alloc((size_t)HIDC * HIDC * 2);
    unsigned short* w_c1r = (unsigned short*)alloc((size_t)HIDC * HIDC * 2);
    unsigned short* w_c2l = (unsigned short*)alloc((size_t)HIDC * HIDC * 2);
    unsigned short* w_c2r = (unsigned short*)alloc((size_t)HIDC * HIDC * 2);
    unsigned short* w_c3l = (unsigned short*)alloc((size_t)HIDC * OUTC * 2);
    unsigned short* w_c3r = (unsigned short*)alloc((size_t)HIDC * OUTC * 2);
    unsigned short* w_hh1 = (unsigned short*)alloc((size_t)HIDC * HIDC * 2);
    unsigned short* w_hh2 = (unsigned short*)alloc((size_t)HIDC * HIDC * 2);
    unsigned short* w_oo  = (unsigned short*)alloc((size_t)OUTC * OUTC * 2);
    unsigned short* w_oh  = (unsigned short*)alloc((size_t)OUTC * HIDC * 2);

    auto packb = [&](const float* w, unsigned short* o, int K, int Nc) {
        int n = K * Nc;
        pack_b_kernel<<<(n + 255) / 256, 256, 0, stream>>>(w, o, K, Nc);
    };
    auto edge = [&](const float* h, const float* we, const float* be) {
        hipMemsetAsync(agg, 0, (size_t)NN * HIDC * 4, stream);
        int total = EE * (HIDC / 4);
        edge_msg_kernel<<<(total + 255) / 256, 256, 0, stream>>>(h, src, dst, ea, we, be,
                                                                 agg, EE, HIDC);
        int n = NN * HIDC;
        rescale_bf_kernel<<<(n + 255) / 256, 256, 0, stream>>>(agg, deg, agg_bf, NN, HIDC);
    };

    // ---- bf16 conversion + WMMA-fragment packing (deterministic every call) ----
    cvt_bf_kernel<<<(NN * INC + 255) / 256, 256, 0, stream>>>(x, x_bf, NN * INC);
    packb(prelin_w, w_prelin, INC, HIDC);
    packb(c1_wl, w_c1l, HIDC, HIDC);  packb(c1_wr, w_c1r, HIDC, HIDC);
    packb(c2_wl, w_c2l, HIDC, HIDC);  packb(c2_wr, w_c2r, HIDC, HIDC);
    packb(c3_wl, w_c3l, HIDC, OUTC);  packb(c3_wr, w_c3r, HIDC, OUTC);
    packb(hh1_w, w_hh1, HIDC, HIDC);  packb(hh2_w, w_hh2, HIDC, HIDC);
    packb(oo_w, w_oo, OUTC, OUTC);    packb(oh_w, w_oh, OUTC, HIDC);

    // ---- in-degree (same dst for all convs) ----
    hipMemsetAsync(deg, 0, (size_t)NN * 4, stream);
    degree_kernel<<<(EE + 255) / 256, 256, 0, stream>>>(dst, deg, EE);

    // ---- prelin: h = relu(x @ W + b) ----
    launch_gemm<INC, 1, true, false, true>(x_bf, w_prelin, prelin_b, nullptr,
                                           hA, hA_bf, NN, HIDC, stream);

    // ---- conv1 + hh1 ----
    edge(hA, c1_we, c1_be);
    launch_gemm<HIDC, 0, true, false, false>(agg_bf, w_c1l, c1_bl, nullptr,
                                             tmp, nullptr, NN, HIDC, stream);
    launch_gemm<HIDC, 1, false, true, true>(hA_bf, w_c1r, nullptr, tmp,
                                            hB, hB_bf, NN, HIDC, stream);
    launch_gemm<HIDC, 2, true, false, true>(hB_bf, w_hh1, hh1_b, nullptr,
                                            hA, hA_bf, NN, HIDC, stream);

    // ---- conv2 + hh2 ----
    edge(hA, c2_we, c2_be);
    launch_gemm<HIDC, 0, true, false, false>(agg_bf, w_c2l, c2_bl, nullptr,
                                             tmp, nullptr, NN, HIDC, stream);
    launch_gemm<HIDC, 1, false, true, true>(hA_bf, w_c2r, nullptr, tmp,
                                            hB, hB_bf, NN, HIDC, stream);
    launch_gemm<HIDC, 2, true, false, true>(hB_bf, w_hh2, hh2_b, nullptr,
                                            hA, hA_bf, NN, HIDC, stream);

    // ---- conv3 (256 -> 128) + oo ----
    edge(hA, c3_we, c3_be);
    launch_gemm<HIDC, 0, true, false, false>(agg_bf, w_c3l, c3_bl, nullptr,
                                             tmp, nullptr, NN, OUTC, stream);
    launch_gemm<HIDC, 1, false, true, true>(hA_bf, w_c3r, nullptr, tmp,
                                            hB, hB_bf, NN, OUTC, stream);
    launch_gemm<OUTC, 2, true, false, false>(hB_bf, w_oo, oo_b, nullptr,
                                             hA, nullptr, NN, OUTC, stream);   // h7 in hA

    // ---- global mean pool -> g[64,128] ----
    hipMemsetAsync(g, 0, (size_t)GG * OUTC * 4, stream);
    hipMemsetAsync(bcnt, 0, (size_t)GG * 4, stream);
    {
        int total = NN * (OUTC / 4);
        pool_sum_kernel<<<(total + 255) / 256, 256, 0, stream>>>(hA, batch, g, bcnt, NN, OUTC);
        int n = GG * OUTC;
        pool_div_kernel<<<(n + 255) / 256, 256, 0, stream>>>(g, bcnt, g_bf, GG, OUTC);
    }

    // ---- head: oh GEMM, batchnorm + leaky, final linear + relu ----
    launch_gemm<OUTC, 0, true, false, false>(g_bf, w_oh, oh_b, nullptr,
                                             g2, nullptr, GG, HIDC, stream);
    bn_leaky_kernel<<<1, HIDC, 0, stream>>>(g2, bn_g, bn_b, GG, HIDC);
    head_out_kernel<<<1, GG, 0, stream>>>(g2, h1_w, h1_b, out, GG, HIDC);
}
